// GravNetLayer_84396107366877
// MI455X (gfx1250) — matrix operand order repaired and checked by reading, not compiled, not run on hardware
//
#include <hip/hip_runtime.h>
#include <hip/hip_bf16.h>

// Problem constants (from reference)
#define B_      128
#define V_      256
#define F_      64
#define NS_     4
#define NLR_    64
#define K_      40
#define NOUT_   128
#define FEFF_   128           // 2*F
#define NSLR_   68            // NS + NLR
#define NPAD_   80            // padded slr column count
#define FIN_    256           // FEFF + 2*NLR
#define MTOT_   (B_ * V_)     // 32768 rows

typedef __attribute__((ext_vector_type(16))) _Float16 v16h;
typedef __attribute__((ext_vector_type(8)))  _Float16 v8h;
typedef __attribute__((ext_vector_type(8)))  float    v8f;

// ---------------------------------------------------------------------------
// WMMA fragment loaders (16x16x32 f16; this code's contiguous k order, same for A and B)
// A 16x32 (MxK), row-major source: lane L -> row M=(L&15); kbase = (L>=16)?16:0;
//   elements 0..7  = K[kbase .. kbase+7]
//   elements 8..15 = K[kbase+8 .. kbase+15]
// B 32x16 (KxN), N-major source (row = N, 32 contiguous K): lane L -> col N=(L&15);
//   kbase = (L>=16)?16:0; elements e = K[kbase+e]  (one contiguous 32B chunk)
// C/D 16x16 f32: element r -> M = r + 8*(L>=16), N = (L&15)
// ---------------------------------------------------------------------------
__device__ __forceinline__ v16h load_a_frag(const _Float16* rowptr, int lane) {
    const int kbase = (lane >> 4) * 16;
    v8h lo = *(const v8h*)(rowptr + kbase);
    v8h hi = *(const v8h*)(rowptr + kbase + 8);
    return __builtin_shufflevector(lo, hi, 0, 1, 2, 3, 4, 5, 6, 7,
                                            8, 9, 10, 11, 12, 13, 14, 15);
}

__device__ __forceinline__ v16h load_b_frag(const _Float16* nrowptr, int lane) {
    const int kbase = (lane >> 4) * 16;
    return *(const v16h*)(nrowptr + kbase);
}

// Async copy `bytes` (multiple of 16*blockDim) from global to LDS, per-lane 16B
// chunks via gfx1250 GLOBAL_LOAD_ASYNC_TO_LDS_B128 (GVS addressing), then wait
// on ASYNCcnt. Caller must __syncthreads() afterwards.
__device__ __forceinline__ void async_stage_to_lds(_Float16* lds_dst,
                                                   const _Float16* gsrc,
                                                   int bytes, int tid, int nthr) {
    // low 32 bits of a generic LDS pointer are the LDS byte offset
    const unsigned lds_base = (unsigned)(size_t)lds_dst;
    for (int c = tid * 16; c < bytes; c += nthr * 16) {
        unsigned lds_addr = lds_base + (unsigned)c;
        unsigned goff     = (unsigned)c;
        asm volatile("global_load_async_to_lds_b128 %0, %1, %2"
                     :: "v"(lds_addr), "v"(goff), "s"(gsrc)
                     : "memory");
    }
    asm volatile("s_wait_asynccnt 0" ::: "memory");
}

// ---------------------------------------------------------------------------
// Kernel 1: per-batch feature mean over V
// ---------------------------------------------------------------------------
__global__ __launch_bounds__(64) void mean_kernel(const float* __restrict__ x,
                                                  float* __restrict__ mean) {
    const int b = blockIdx.x;
    const int f = threadIdx.x;
    const float* xp = x + (size_t)b * V_ * F_ + f;
    float sum = 0.f;
    #pragma unroll 8
    for (int v = 0; v < V_; ++v) sum += xp[v * F_];
    mean[b * F_ + f] = sum * (1.0f / V_);
}

// ---------------------------------------------------------------------------
// Kernel 2: build f16 x_cat  ->  Ah (32768x128) and fp_h[:, 0:128]
// ---------------------------------------------------------------------------
__global__ __launch_bounds__(256) void packA_kernel(const float* __restrict__ x,
                                                    const float* __restrict__ mean,
                                                    _Float16* __restrict__ Ah,
                                                    _Float16* __restrict__ fph) {
    const int idx = blockIdx.x * 256 + threadIdx.x;   // over 32768*128
    const int row = idx >> 7;
    const int k   = idx & 127;
    const int b   = row >> 8;
    float val = (k < F_) ? x[(size_t)row * F_ + k] : mean[b * F_ + (k - F_)];
    _Float16 h = (_Float16)val;
    Ah[idx] = h;
    fph[(size_t)row * FIN_ + k] = h;
}

// ---------------------------------------------------------------------------
// Kernel 3: transpose weights to N-major f16 (zero-pad W_slr cols to 80)
// ---------------------------------------------------------------------------
__global__ __launch_bounds__(256) void packW_kernel(const float* __restrict__ W_slr,
                                                    const float* __restrict__ W_out,
                                                    _Float16* __restrict__ Wt1,   // 80 x 128
                                                    _Float16* __restrict__ Wt2) { // 128 x 256
    const int idx = blockIdx.x * 256 + threadIdx.x;
    const int n1 = NPAD_ * FEFF_;       // 10240
    const int n2 = NOUT_ * FIN_;        // 32768
    if (idx < n1) {
        const int n = idx / FEFF_;
        const int k = idx % FEFF_;
        Wt1[idx] = (n < NSLR_) ? (_Float16)W_slr[k * NSLR_ + n] : (_Float16)0.f;
    } else if (idx < n1 + n2) {
        const int j = idx - n1;
        const int n = j / FIN_;
        const int k = j % FIN_;
        Wt2[j] = (_Float16)W_out[k * NOUT_ + n];
    }
}

// ---------------------------------------------------------------------------
// Kernel 4: GEMM1  slr = relu(x_cat @ W_slr + b) -> padded fp32 slr (M x 80)
// block = 128 (4 waves), each wave owns a 16-row M-tile; B staged in LDS via
// async copy. Branch-free padded epilogue (cols 68..79 are zero-weight junk).
// ---------------------------------------------------------------------------
__global__ __launch_bounds__(128) void gemm1_kernel(const _Float16* __restrict__ Ah,
                                                    const _Float16* __restrict__ Wt1,
                                                    const float* __restrict__ b_slr,
                                                    float* __restrict__ slr_f) {
    __shared__ _Float16 Bs[NPAD_ * FEFF_];            // 20 KB
    const int tid   = threadIdx.x;
    const int lane  = tid & 31;
    const int wave  = tid >> 5;
    const int tileM = blockIdx.x * 64 + wave * 16;
    const int nloc  = lane & 15;
    const int roff  = (lane >> 4) * 8;

    async_stage_to_lds(Bs, Wt1, NPAD_ * FEFF_ * 2, tid, 128);

    const _Float16* arow = Ah + (size_t)(tileM + nloc) * FEFF_;
    v16h afrag[4];
    #pragma unroll
    for (int ks = 0; ks < 4; ++ks) afrag[ks] = load_a_frag(arow + ks * 32, lane);

    __syncthreads();   // Bs visible to all waves

    #pragma unroll
    for (int nt = 0; nt < 5; ++nt) {
        v8f acc = {};
        const _Float16* brow = Bs + (nt * 16 + nloc) * FEFF_;
        #pragma unroll
        for (int ks = 0; ks < 4; ++ks) {
            v16h bfrag = load_b_frag(brow + ks * 32, lane);
            acc = __builtin_amdgcn_wmma_f32_16x16x32_f16(
                false, afrag[ks], false, bfrag, (short)0, acc, false, false);
        }
        const int col = nt * 16 + nloc;
        const float bias = (col < NSLR_) ? b_slr[col] : 0.f;
        #pragma unroll
        for (int r = 0; r < 8; ++r) {
            float val = acc[r] + bias;
            val = val > 0.f ? val : 0.f;
            slr_f[(size_t)(tileM + roff + r) * NPAD_ + col] = val;
        }
    }
}

// ---------------------------------------------------------------------------
// Kernel 5: kNN + weighted mean/max aggregation.
// One block (256 threads) per node. Rank-select: rank(j) = #{l : (d2,l) < (d2,j)}
// (strict lexicographic order -> unique ranks, same selected set as top_k).
// slr_f row layout: [0..3]=s coords, [4..67]=lr features, [68..79]=pad.
// ---------------------------------------------------------------------------
__global__ __launch_bounds__(256) void knn_kernel(const float* __restrict__ slr_f,
                                                  _Float16* __restrict__ fph) {
    const int node  = blockIdx.x;       // b*256 + i
    const int b     = node >> 8;
    const int tid   = threadIdx.x;
    const int bbase = b * V_;

    __shared__ float si[NS_];
    __shared__ float d2s[V_];
    __shared__ int   selj[K_];
    __shared__ float w_s[K_];
    __shared__ float psum[4][NLR_];
    __shared__ float pmax[4][NLR_];

    if (tid < NS_) si[tid] = slr_f[(size_t)node * NPAD_ + tid];
    __syncthreads();

    const float* sj = slr_f + (size_t)(bbase + tid) * NPAD_;
    float d2 = 0.f;
    #pragma unroll
    for (int c = 0; c < NS_; ++c) {
        const float diff = si[c] - sj[c];
        d2 += diff * diff;
    }
    d2s[tid] = d2;
    __syncthreads();

    int rank = 0;
    for (int l = 0; l < V_; ++l) {
        const float dl = d2s[l];
        rank += (dl < d2) || (dl == d2 && l < tid);
    }
    if (rank < K_) {
        selj[rank] = tid;
        w_s[rank]  = __expf(-10.0f * d2);
    }
    __syncthreads();

    // weighted features: 4 groups of 64 threads split the K=40 neighbors
    const int g = tid >> 6;
    const int f = tid & 63;
    float sum = 0.f, mx = -1e30f;
    for (int k = g; k < K_; k += 4) {
        const int jj  = selj[k];
        const float w = w_s[k];
        const float val = slr_f[(size_t)(bbase + jj) * NPAD_ + NS_ + f] * w;
        sum += val;
        mx = fmaxf(mx, val);
    }
    psum[g][f] = sum;
    pmax[g][f] = mx;
    __syncthreads();

    if (tid < NLR_) {
        const float tot = psum[0][f] + psum[1][f] + psum[2][f] + psum[3][f];
        const float m   = fmaxf(fmaxf(pmax[0][f], pmax[1][f]),
                                fmaxf(pmax[2][f], pmax[3][f]));
        _Float16* fr = fph + (size_t)node * FIN_;
        fr[FEFF_ + f]        = (_Float16)(tot * (1.0f / K_));   // ft.mean
        fr[FEFF_ + NLR_ + f] = (_Float16)m;                     // ft.max
    }
}

// ---------------------------------------------------------------------------
// Kernel 6: GEMM2  out = relu(fp @ W_out + b_out), fp32 output
// block = 128 (4 waves); full 64 KB weight panel staged in LDS via async copy;
// 8 N-tiles x 8 K-steps of WMMA per wave.
// ---------------------------------------------------------------------------
__global__ __launch_bounds__(128) void gemm2_kernel(const _Float16* __restrict__ fph,
                                                    const _Float16* __restrict__ Wt2,
                                                    const float* __restrict__ b_out,
                                                    float* __restrict__ out) {
    __shared__ _Float16 Bs[NOUT_ * FIN_];             // 64 KB
    const int tid   = threadIdx.x;
    const int lane  = tid & 31;
    const int wave  = tid >> 5;
    const int tileM = blockIdx.x * 64 + wave * 16;
    const int nloc  = lane & 15;
    const int roff  = (lane >> 4) * 8;

    async_stage_to_lds(Bs, Wt2, NOUT_ * FIN_ * 2, tid, 128);

    const _Float16* arow = fph + (size_t)(tileM + nloc) * FIN_;
    v16h afrag[8];
    #pragma unroll
    for (int ks = 0; ks < 8; ++ks) afrag[ks] = load_a_frag(arow + ks * 32, lane);

    __syncthreads();   // Bs visible to all waves

    #pragma unroll
    for (int nt = 0; nt < 8; ++nt) {
        v8f acc = {};
        const _Float16* brow = Bs + (nt * 16 + nloc) * FIN_;
        #pragma unroll
        for (int ks = 0; ks < 8; ++ks) {
            v16h bfrag = load_b_frag(brow + ks * 32, lane);
            acc = __builtin_amdgcn_wmma_f32_16x16x32_f16(
                false, afrag[ks], false, bfrag, (short)0, acc, false, false);
        }
        const int col = nt * 16 + nloc;
        const float bias = b_out[col];
        #pragma unroll
        for (int r = 0; r < 8; ++r) {
            float val = acc[r] + bias;
            val = val > 0.f ? val : 0.f;
            out[(size_t)(tileM + roff + r) * NOUT_ + col] = val;
        }
    }
}

// ---------------------------------------------------------------------------
// Workspace layout (bytes, 256B aligned):
//   mean   [0,        32768)     B*64 f32
//   slr_f  [32768,    10518528)  32768*80 f32 (padded s|lr)
//   Ah     [10518528, 18907136)  32768*128 f16
//   fp_h   [18907136, 35684352)  32768*256 f16
//   Wt1    [35684352, 35704832)  80*128 f16
//   Wt2    [35704832, 35770368)  128*256 f16
// ---------------------------------------------------------------------------
extern "C" void kernel_launch(void* const* d_in, const int* in_sizes, int n_in,
                              void* d_out, int out_size, void* d_ws, size_t ws_size,
                              hipStream_t stream) {
    const float* x     = (const float*)d_in[0];
    const float* W_slr = (const float*)d_in[1];
    const float* b_slr = (const float*)d_in[2];
    const float* W_out = (const float*)d_in[3];
    const float* b_out = (const float*)d_in[4];
    float* out = (float*)d_out;

    char* ws = (char*)d_ws;
    float*    mean  = (float*)(ws + 0);
    float*    slr_f = (float*)(ws + 32768);
    _Float16* Ah    = (_Float16*)(ws + 10518528);
    _Float16* fph   = (_Float16*)(ws + 18907136);
    _Float16* Wt1   = (_Float16*)(ws + 35684352);
    _Float16* Wt2   = (_Float16*)(ws + 35704832);

    mean_kernel<<<B_, 64, 0, stream>>>(x, mean);
    packA_kernel<<<(MTOT_ * FEFF_) / 256, 256, 0, stream>>>(x, mean, Ah, fph);
    packW_kernel<<<(NPAD_ * FEFF_ + NOUT_ * FIN_ + 255) / 256, 256, 0, stream>>>(
        W_slr, W_out, Wt1, Wt2);
    gemm1_kernel<<<MTOT_ / 64, 128, 0, stream>>>(Ah, Wt1, b_slr, slr_f);
    knn_kernel<<<MTOT_, 256, 0, stream>>>(slr_f, fph);
    gemm2_kernel<<<MTOT_ / 64, 128, 0, stream>>>(fph, Wt2, b_out, out);
}
